// MultiHeadAttention_15891378995789
// MI455X (gfx1250) — matrix operand order repaired and checked by reading, not compiled, not run on hardware
//
#include <hip/hip_runtime.h>

// ---------------------------------------------------------------------------
// MHA forward for B=4, S=2048, E=1024, H=16, DH=64 on gfx1250 (MI455X).
// bf16 WMMA (v_wmma_f32_16x16x32_bf16) everywhere, f32 accumulate.
// ---------------------------------------------------------------------------

namespace {
constexpr int kB = 4, kS = 2048, kE = 1024, kH = 16, kD = 64;
constexpr int kM = kB * kS;                                   // 8192 token rows
constexpr float kQScale = 0.125f * 1.44269504088896341f;      // DH^-0.5 * log2(e)
}

typedef __bf16 bf16_t;
typedef __attribute__((ext_vector_type(8)))  __bf16 bf16x8;
typedef __attribute__((ext_vector_type(16))) __bf16 bf16x16;
typedef __attribute__((ext_vector_type(8)))  float  f32x8;

union Frag16 { bf16x16 v; bf16x8 h[2]; };

// A-matrix 16x32 bf16 fragment (M x K), source row-major with leading dim ld.
// Lane l: row = l&15; K set = {kb..kb+7, kb+16..kb+23}, kb = (l<16)?0:8.
static __device__ __forceinline__ bf16x16 loadAfrag(const bf16_t* __restrict__ base, int ld) {
  const int lane = threadIdx.x & 31;
  const int row  = lane & 15;
  const int kb   = (lane >> 4) << 3;            // 0 or 8
  const bf16_t* p = base + row * ld + kb;
  Frag16 f;
  f.h[0] = *(const bf16x8*)(p);
  f.h[1] = *(const bf16x8*)(p + 16);
  return f.v;
}

// B-matrix 32x16 bf16 fragment (K x N), with memory holding B^T row-major:
// column n's 32 K-values are contiguous at base + n*ld. Lane l: col = l&15,
// K set = {kb..kb+7, kb+16..kb+23}, kb = (l<16)?0:8.
static __device__ __forceinline__ bf16x16 loadBfrag(const bf16_t* __restrict__ base, long ld) {
  const int lane = threadIdx.x & 31;
  const int col  = lane & 15;
  const int kb   = (lane >> 4) << 3;            // 0 or 8
  const bf16_t* p = base + (long)col * ld + kb;
  Frag16 f;
  f.h[0] = *(const bf16x8*)(p);
  f.h[1] = *(const bf16x8*)(p + 16);
  return f.v;
}

static __device__ __forceinline__ f32x8 wmma_bf16(bf16x16 a, bf16x16 b, f32x8 c) {
  return __builtin_amdgcn_wmma_f32_16x16x32_bf16(false, a, false, b, (short)0, c,
                                                 false, false);
}

// ---------------------------------------------------------------------------
__global__ void cast_f32_to_bf16(const float* __restrict__ in,
                                 bf16_t* __restrict__ out, int n) {
  int i = blockIdx.x * blockDim.x + threadIdx.x;
  if (i < n) out[i] = (bf16_t)in[i];
}

// ---------------------------------------------------------------------------
// QKV projection: [8192,1024] x [3072,1024]^T. Block tile 128x128 (8 waves,
// each wave 64x32). Epilogue scatters into Q (scaled), K, and V^T layouts.
// grid = (3E/128, M/128) = (24, 64), 256 threads.
__global__ __launch_bounds__(256) void qkv_gemm_kernel(
    const bf16_t* __restrict__ X, const bf16_t* __restrict__ W,
    bf16_t* __restrict__ Qm, bf16_t* __restrict__ Km, bf16_t* __restrict__ Vt) {
  const int lane  = threadIdx.x & 31;
  const int wid   = threadIdx.x >> 5;
  const int mwave = wid >> 2;                    // 0..1
  const int nwave = wid & 3;                     // 0..3
  const int m0 = blockIdx.y * 128 + mwave * 64;
  const int n0 = blockIdx.x * 128 + nwave * 32;

  f32x8 acc[4][2];
#pragma unroll
  for (int i = 0; i < 4; ++i)
#pragma unroll
    for (int j = 0; j < 2; ++j)
      acc[i][j] = f32x8{0.f, 0.f, 0.f, 0.f, 0.f, 0.f, 0.f, 0.f};

#pragma unroll 2
  for (int k0 = 0; k0 < kE; k0 += 32) {
    bf16x16 a[4], bfr[2];
#pragma unroll
    for (int i = 0; i < 4; ++i)
      a[i] = loadAfrag(X + (long)(m0 + 16 * i) * kE + k0, kE);
#pragma unroll
    for (int j = 0; j < 2; ++j)
      bfr[j] = loadBfrag(W + (long)(n0 + 16 * j) * kE + k0, kE);
#pragma unroll
    for (int i = 0; i < 4; ++i)
#pragma unroll
      for (int j = 0; j < 2; ++j)
        acc[i][j] = wmma_bf16(a[i], bfr[j], acc[i][j]);
  }

  // Epilogue: D element (vgpr v, lane l) -> row m0+16i+v+8*(l>=16), col n0+16j+(l&15).
  const int hf = lane >> 4;
  const int c  = lane & 15;
#pragma unroll
  for (int i = 0; i < 4; ++i) {
#pragma unroll
    for (int j = 0; j < 2; ++j) {
      const int n  = n0 + 16 * j + c;
      const int kk = n >> 10;                    // 0=Q 1=K 2=V
      const int d  = (n >> 4) & 63;
      const int h  = n & 15;
#pragma unroll
      for (int v = 0; v < 8; ++v) {
        const int m  = m0 + 16 * i + v + 8 * hf;
        const int bb = m >> 11;                  // m / S
        const int s  = m & (kS - 1);
        const float val = acc[i][j][v];
        const long bh = (long)bb * kH + h;
        if (kk == 0) {
          Qm[(bh * kS + s) * kD + d] = (bf16_t)(val * kQScale);
        } else if (kk == 1) {
          Km[(bh * kS + s) * kD + d] = (bf16_t)val;
        } else {
          Vt[(bh * kD + d) * (long)kS + s] = (bf16_t)val;   // V transposed
        }
      }
    }
  }
}

// ---------------------------------------------------------------------------
// Flash attention. grid = (S/128, B*H) = (16, 64), 256 threads.
// Each wave owns 16 query rows; streams K/V in 32-key chunks.
__global__ __launch_bounds__(256) void attn_kernel(
    const bf16_t* __restrict__ Qm, const bf16_t* __restrict__ Km,
    const bf16_t* __restrict__ Vt, bf16_t* __restrict__ Ctx) {
  __shared__ __align__(64) bf16_t plds[8][16][32];   // per-wave P staging

  const int lane = threadIdx.x & 31;
  const int wid  = threadIdx.x >> 5;
  const int bh   = blockIdx.y;                   // b*H + h
  const int b    = bh >> 4;
  const int h    = bh & 15;
  const int q0   = blockIdx.x * 128 + wid * 16;

  const bf16_t* qptr = Qm + ((long)bh * kS + q0) * kD;
  const bf16_t* kptr = Km + (long)bh * kS * kD;
  const bf16_t* vtp  = Vt + (long)bh * kD * kS;

  const bf16x16 qa0 = loadAfrag(qptr, kD);       // d = 0..31
  const bf16x16 qa1 = loadAfrag(qptr + 32, kD);  // d = 32..63

  f32x8 o[4];
#pragma unroll
  for (int t = 0; t < 4; ++t) o[t] = f32x8{0.f, 0.f, 0.f, 0.f, 0.f, 0.f, 0.f, 0.f};
  float mrun[8], lsum[8];
#pragma unroll
  for (int v = 0; v < 8; ++v) { mrun[v] = -1e30f; lsum[v] = 0.f; }

  const int hf = lane >> 4;
  const int c  = lane & 15;

  for (int j0 = 0; j0 < kS; j0 += 32) {
    // ---- scores: two 16x16 tiles over keys [j0, j0+32) --------------------
    f32x8 s0 = f32x8{0.f, 0.f, 0.f, 0.f, 0.f, 0.f, 0.f, 0.f};
    f32x8 s1 = s0;
    const bf16x16 kb00 = loadBfrag(kptr + (long)j0 * kD, kD);
    const bf16x16 kb01 = loadBfrag(kptr + (long)j0 * kD + 32, kD);
    const bf16x16 kb10 = loadBfrag(kptr + (long)(j0 + 16) * kD, kD);
    const bf16x16 kb11 = loadBfrag(kptr + (long)(j0 + 16) * kD + 32, kD);
    s0 = wmma_bf16(qa0, kb00, s0);
    s0 = wmma_bf16(qa1, kb01, s0);
    s1 = wmma_bf16(qa0, kb10, s1);
    s1 = wmma_bf16(qa1, kb11, s1);

    // ---- online softmax (rows live in 16-lane halves) ---------------------
    float corr[8];
#pragma unroll
    for (int v = 0; v < 8; ++v) {
      const float a = s0[v];
      const float e = s1[v];
      float mx = fmaxf(a, e);
      mx = fmaxf(mx, __shfl_xor(mx, 1));
      mx = fmaxf(mx, __shfl_xor(mx, 2));
      mx = fmaxf(mx, __shfl_xor(mx, 4));
      mx = fmaxf(mx, __shfl_xor(mx, 8));
      const float mnew = fmaxf(mrun[v], mx);
      const float cf = __builtin_amdgcn_exp2f(mrun[v] - mnew);
      mrun[v] = mnew;
      const float p0 = __builtin_amdgcn_exp2f(a - mnew);
      const float p1 = __builtin_amdgcn_exp2f(e - mnew);
      float rs = p0 + p1;
      rs += __shfl_xor(rs, 1);
      rs += __shfl_xor(rs, 2);
      rs += __shfl_xor(rs, 4);
      rs += __shfl_xor(rs, 8);
      lsum[v] = lsum[v] * cf + rs;
      corr[v] = cf;
      // stage P row-major [16][32] in wave-private LDS (C/D -> A relayout)
      plds[wid][v + 8 * hf][c]      = (bf16_t)p0;
      plds[wid][v + 8 * hf][c + 16] = (bf16_t)p1;
    }
    // same-wave LDS ops are in-order; wait DScnt=0 before re-reading P.
    asm volatile("s_wait_dscnt 0" ::: "memory");
    const bf16x16 pa = loadAfrag(&plds[wid][0][0], 32);

    // ---- rescale accumulators, then PV ------------------------------------
#pragma unroll
    for (int t = 0; t < 4; ++t)
#pragma unroll
      for (int v = 0; v < 8; ++v) o[t][v] *= corr[v];
#pragma unroll
    for (int t = 0; t < 4; ++t) {
      const bf16x16 vb = loadBfrag(vtp + (long)(t * 16) * kS + j0, kS);
      o[t] = wmma_bf16(pa, vb, o[t]);
    }
  }

  // ---- normalize and write ctx[b, s, h*64 + d] ----------------------------
#pragma unroll
  for (int t = 0; t < 4; ++t) {
#pragma unroll
    for (int v = 0; v < 8; ++v) {
      const int r = v + 8 * hf;
      const int d = t * 16 + c;
      Ctx[((long)(b * kS + q0 + r)) * kE + h * kD + d] = (bf16_t)(o[t][v] / lsum[v]);
    }
  }
}

// ---------------------------------------------------------------------------
// Output projection: [8192,1024] x [1024,1024]^T + bias, fp32 out.
// grid = (E/128, M/128) = (8, 64), 256 threads.
__global__ __launch_bounds__(256) void out_gemm_kernel(
    const bf16_t* __restrict__ Cx, const bf16_t* __restrict__ W,
    const float* __restrict__ bias, float* __restrict__ Out) {
  const int lane  = threadIdx.x & 31;
  const int wid   = threadIdx.x >> 5;
  const int mwave = wid >> 2;
  const int nwave = wid & 3;
  const int m0 = blockIdx.y * 128 + mwave * 64;
  const int n0 = blockIdx.x * 128 + nwave * 32;

  f32x8 acc[4][2];
#pragma unroll
  for (int i = 0; i < 4; ++i)
#pragma unroll
    for (int j = 0; j < 2; ++j)
      acc[i][j] = f32x8{0.f, 0.f, 0.f, 0.f, 0.f, 0.f, 0.f, 0.f};

#pragma unroll 2
  for (int k0 = 0; k0 < kE; k0 += 32) {
    bf16x16 a[4], bfr[2];
#pragma unroll
    for (int i = 0; i < 4; ++i)
      a[i] = loadAfrag(Cx + (long)(m0 + 16 * i) * kE + k0, kE);
#pragma unroll
    for (int j = 0; j < 2; ++j)
      bfr[j] = loadBfrag(W + (long)(n0 + 16 * j) * kE + k0, kE);
#pragma unroll
    for (int i = 0; i < 4; ++i)
#pragma unroll
      for (int j = 0; j < 2; ++j)
        acc[i][j] = wmma_bf16(a[i], bfr[j], acc[i][j]);
  }

  const int hf = lane >> 4;
  const int c  = lane & 15;
#pragma unroll
  for (int i = 0; i < 4; ++i) {
#pragma unroll
    for (int j = 0; j < 2; ++j) {
      const int n = n0 + 16 * j + c;
      const float bv = bias[n];
#pragma unroll
      for (int v = 0; v < 8; ++v) {
        const int m = m0 + 16 * i + v + 8 * hf;
        Out[(long)m * kE + n] = acc[i][j][v] + bv;
      }
    }
  }
}

// ---------------------------------------------------------------------------
extern "C" void kernel_launch(void* const* d_in, const int* in_sizes, int n_in,
                              void* d_out, int out_size, void* d_ws, size_t ws_size,
                              hipStream_t stream) {
  (void)in_sizes; (void)n_in; (void)out_size; (void)ws_size;
  const float* x    = (const float*)d_in[0];   // [B,S,E]
  const float* wqkv = (const float*)d_in[1];   // [3E,E]
  const float* wout = (const float*)d_in[2];   // [E,E]
  const float* bout = (const float*)d_in[3];   // [E]
  float* out = (float*)d_out;                  // [B,S,E] fp32

  // Workspace layout (bf16 elements): 88 MiB total.
  bf16_t* ws  = (bf16_t*)d_ws;
  bf16_t* xb  = ws;                              // M*E          = 8388608
  bf16_t* wqb = xb  + (size_t)kM * kE;           // 3E*E         = 3145728
  bf16_t* wob = wqb + (size_t)3 * kE * kE;       // E*E          = 1048576
  bf16_t* qm  = wob + (size_t)kE * kE;           // B*H*S*DH     = 8388608
  bf16_t* km  = qm  + (size_t)kB * kH * kS * kD;
  bf16_t* vt  = km  + (size_t)kB * kH * kS * kD;
  bf16_t* ctx = vt  + (size_t)kB * kH * kS * kD; // M*E          = 8388608

  const int nX = kM * kE, nWq = 3 * kE * kE, nWo = kE * kE;
  cast_f32_to_bf16<<<(nX  + 255) / 256, 256, 0, stream>>>(x,    xb,  nX);
  cast_f32_to_bf16<<<(nWq + 255) / 256, 256, 0, stream>>>(wqkv, wqb, nWq);
  cast_f32_to_bf16<<<(nWo + 255) / 256, 256, 0, stream>>>(wout, wob, nWo);

  qkv_gemm_kernel<<<dim3(3 * kE / 128, kM / 128), 256, 0, stream>>>(xb, wqb, qm, km, vt);
  attn_kernel<<<dim3(kS / 128, kB * kH), 256, 0, stream>>>(qm, km, vt, ctx);
  out_gemm_kernel<<<dim3(kE / 128, kM / 128), 256, 0, stream>>>(ctx, wob, bout, out);
}
